// MultiScaleROISegmentationHead_24790551233451
// MI455X (gfx1250) — matrix operand order repaired and checked by reading, not compiled, not run on hardware
//
#include <hip/hip_runtime.h>

typedef __attribute__((ext_vector_type(16))) _Float16 v16h;
typedef __attribute__((ext_vector_type(8)))  _Float16 v8h;
typedef __attribute__((ext_vector_type(8)))  float    v8f;

union AF { v16h v; v8h h[2]; unsigned u[16]; };

#define P28  37632    // 48*28*28
#define P56  150528   // 48*56*56
#define P112 602112   // 48*112*112

// ---------------- workspace layout (bytes) ----------------
static const size_t OFF_ACC   = 0;                                   // f32, P112*128
static const size_t OFF_X0    = (size_t)P112 * 128 * 4;              // f16, P112*128
static const size_t OFF_X1    = OFF_X0 + (size_t)P112 * 128 * 2;     // f16, P112*128
static const size_t OFF_FUSED = OFF_X1 + (size_t)P112 * 128 * 2;     // f32, P28*256
static const size_t OFF_W     = OFF_FUSED + (size_t)P28 * 256 * 4;   // f16 weights
static const size_t OFF_FW    = OFF_W + (size_t)4415488 * 2;         // 4 floats

// f16 weight sub-offsets (in halves)
static const size_t WOFF_RED = 0;        // 4 x 65536   [O][C]
static const size_t WOFF_RES = 262144;   // 6 x 589824  [t][O][C]
static const size_t WOFF_UP1 = 3801088;  // 262144      [ab][D][C]
static const size_t WOFF_UP2 = 4063232;  // 131072      [ab][D][C]
static const size_t WOFF_C1  = 4194304;  // 147456      [t][O][C]
static const size_t WOFF_C2  = 4341760;  // 73728       [t][O][C]

// ---------------- weight prep ----------------
__global__ void k_f32_to_f16(const float* __restrict__ s, _Float16* __restrict__ d, int n) {
  int i = blockIdx.x * 256 + threadIdx.x;
  if (i < n) d[i] = (_Float16)s[i];
}

// src OIHW (O,C,ks,ks) -> dst [t][O][C]
__global__ void k_oihw_to_toc(const float* __restrict__ s, _Float16* __restrict__ d,
                              unsigned O, unsigned C, unsigned T) {
  unsigned i = blockIdx.x * 256 + threadIdx.x;
  unsigned n = O * C * T;
  if (i >= n) return;
  unsigned t = i / (O * C); unsigned r = i % (O * C); unsigned o = r / C; unsigned c = r % C;
  d[i] = (_Float16)s[(size_t)(o * C + c) * T + t];
}

// src (C,D,2,2) -> dst [ab][D][C]
__global__ void k_cdab_to_abdc(const float* __restrict__ s, _Float16* __restrict__ d,
                               unsigned C, unsigned D) {
  unsigned i = blockIdx.x * 256 + threadIdx.x;
  unsigned n = 4 * D * C;
  if (i >= n) return;
  unsigned ab = i / (D * C); unsigned r = i % (D * C); unsigned dd = r / C; unsigned c = r % C;
  d[i] = (_Float16)s[(size_t)(c * D + dd) * 4 + ab];
}

__global__ void k_softmax4(const float* __restrict__ w, float* __restrict__ o) {
  if (threadIdx.x == 0 && blockIdx.x == 0) {
    float m = fmaxf(fmaxf(w[0], w[1]), fmaxf(w[2], w[3]));
    float e0 = expf(w[0]-m), e1 = expf(w[1]-m), e2 = expf(w[2]-m), e3 = expf(w[3]-m);
    float s = e0 + e1 + e2 + e3;
    o[0] = e0/s; o[1] = e1/s; o[2] = e2/s; o[3] = e3/s;
  }
}

// ---------------- ROI align (per-level) ----------------
// block = one output pixel (roi, ph, pw); thread = channel. dst f16 NHWC [pix][256]
__global__ void __launch_bounds__(256) k_roi_align(
    const float* __restrict__ feat, const float* __restrict__ rois,
    _Float16* __restrict__ dst, int C, int H, int W, float scale)
{
  unsigned blk = blockIdx.x;            // roi*784 + ph*28 + pw
  int c = threadIdx.x;
  int pw = blk % 28u; unsigned t = blk / 28u; int ph = t % 28u; int roi = t / 28u;
  const float* rr = rois + (size_t)roi * 5;
  int bi = (int)rr[0];
  float x1 = rr[1]*scale - 0.5f, y1 = rr[2]*scale - 0.5f;
  float x2 = rr[3]*scale - 0.5f, y2 = rr[4]*scale - 0.5f;
  float bw = (x2 - x1) * (1.0f/28.0f), bh = (y2 - y1) * (1.0f/28.0f);
  const float* fb = feat + ((size_t)bi * C + c) * H * W;
  float accv = 0.f;
  for (int iy = 0; iy < 2; ++iy) {
    for (int ix = 0; ix < 2; ++ix) {
      float y = y1 + ((float)ph + (iy + 0.5f) * 0.5f) * bh;
      float x = x1 + ((float)pw + (ix + 0.5f) * 0.5f) * bw;
      bool valid = (y > -1.f) && (y < (float)H) && (x > -1.f) && (x < (float)W);
      y = fminf(fmaxf(y, 0.f), (float)H - 1.f);
      x = fminf(fmaxf(x, 0.f), (float)W - 1.f);
      float y0f = fminf(fmaxf(floorf(y), 0.f), (float)(H - 2));
      float x0f = fminf(fmaxf(floorf(x), 0.f), (float)(W - 2));
      int y0 = (int)y0f, x0 = (int)x0f;
      float ly = y - y0f, lx = x - x0f, hy = 1.f - ly, hx = 1.f - lx;
      float v = fb[(size_t)y0*W + x0]     * hy*hx + fb[(size_t)y0*W + x0 + 1]     * hy*lx
              + fb[(size_t)(y0+1)*W + x0] * ly*hx + fb[(size_t)(y0+1)*W + x0 + 1] * ly*lx;
      accv += valid ? v : 0.f;
    }
  }
  dst[(size_t)blk * C + c] = (_Float16)(accv * 0.25f);
}

// ---------------- generic conv (1x1 / 3x3 s1 / 3x3 s2) via WMMA ----------------
// act: f16 NHWC [Nb*Hin*Win][Cin]; wt: f16 [t][Cout][Cin]; acc: f32 [Nb*Hout*Wout][Cout]
// wave tile: 32 out pixels x 64 out channels (grid.y = out-channel tile of 64)
__global__ void __launch_bounds__(256) k_conv_wmma(
    const _Float16* __restrict__ act, const _Float16* __restrict__ wt,
    float* __restrict__ acc,
    int Nb, int Hin, int Win, int Cin, int Hout, int Wout, int Cout,
    int ks, int stride, int pad)
{
  const int lane = threadIdx.x & 31;
  const int wave = threadIdx.x >> 5;
  const unsigned numM = ((unsigned)(Nb * Hout * Wout)) >> 5;   // 32-pixel tiles
  const unsigned mt = blockIdx.x * 8u + (unsigned)wave;
  if (mt >= numM) return;
  const int nb = (int)(blockIdx.y << 6);

  const int col  = lane & 15;
  const int khbA = (lane & 16) ? 8 : 0;    // A: K sub-block per lane half
  const int kbB  = (lane & 16) ? 8 : 0;    // B: K sub-block per lane half

  const unsigned HWo = (unsigned)(Hout * Wout);
  const unsigned pix0 = mt * 32u + (unsigned)col;   // A rows for fragment 0
  const unsigned pix1 = pix0 + 16u;                 // A rows for fragment 1
  const unsigned n0 = pix0 / HWo, r0 = pix0 % HWo;
  const unsigned n1 = pix1 / HWo, r1 = pix1 % HWo;
  const int oy0 = (int)(r0 / (unsigned)Wout), ox0 = (int)(r0 % (unsigned)Wout);
  const int oy1 = (int)(r1 / (unsigned)Wout), ox1 = (int)(r1 % (unsigned)Wout);

  const v8f zf = {0.f,0.f,0.f,0.f,0.f,0.f,0.f,0.f};
  v8f cacc[2][4];
#pragma unroll
  for (int g = 0; g < 2; ++g)
#pragma unroll
    for (int j = 0; j < 4; ++j) cacc[g][j] = zf;

  const int ksteps = Cin >> 5;
  const int T = ks * ks;

  for (int t = 0; t < T; ++t) {
    const int dy = t / ks, dx = t % ks;
    const int sy0 = oy0 * stride + dy - pad, sx0 = ox0 * stride + dx - pad;
    const int sy1 = oy1 * stride + dy - pad, sx1 = ox1 * stride + dx - pad;
    const bool valid0 = (sy0 >= 0) && (sy0 < Hin) && (sx0 >= 0) && (sx0 < Win);
    const bool valid1 = (sy1 >= 0) && (sy1 < Hin) && (sx1 >= 0) && (sx1 < Win);
    const _Float16* ap0 = act + ((size_t)((int)n0 * Hin + sy0) * Win + sx0) * Cin + khbA;
    const _Float16* ap1 = act + ((size_t)((int)n1 * Hin + sy1) * Win + sx1) * Cin + khbA;
    const _Float16* wp  = wt + ((size_t)t * Cout + nb + col) * Cin + kbB;
    for (int kk = 0; kk < ksteps; ++kk) {
      const int k0 = kk << 5;
      AF a0, a1;
#pragma unroll
      for (int i = 0; i < 16; ++i) { a0.u[i] = 0u; a1.u[i] = 0u; }
      if (valid0) { a0.h[0] = *(const v8h*)(ap0 + k0); a0.h[1] = *(const v8h*)(ap0 + k0 + 16); }
      if (valid1) { a1.h[0] = *(const v8h*)(ap1 + k0); a1.h[1] = *(const v8h*)(ap1 + k0 + 16); }
      AF b[4];
#pragma unroll
      for (int j = 0; j < 4; ++j) {
        const _Float16* w0 = wp + (size_t)(j * 16) * Cin + k0;
        b[j].h[0] = *(const v8h*)(w0);
        b[j].h[1] = *(const v8h*)(w0 + 16);
      }
#pragma unroll
      for (int j = 0; j < 4; ++j) {
        cacc[0][j] = __builtin_amdgcn_wmma_f32_16x16x32_f16(false, a0.v, false, b[j].v, (short)0, cacc[0][j], false, false);
        cacc[1][j] = __builtin_amdgcn_wmma_f32_16x16x32_f16(false, a1.v, false, b[j].v, (short)0, cacc[1][j], false, false);
      }
    }
  }
  const int chBase = nb + col;
  const int mAdd = (lane & 16) ? 8 : 0;
#pragma unroll
  for (int g = 0; g < 2; ++g) {
#pragma unroll
    for (int vv = 0; vv < 8; ++vv) {
      const size_t o = (size_t)(mt * 32u + (unsigned)(g * 16 + vv + mAdd)) * Cout + chBase;
      acc[o]      = cacc[g][0][vv];
      acc[o + 16] = cacc[g][1][vv];
      acc[o + 32] = cacc[g][2][vv];
      acc[o + 48] = cacc[g][3][vv];
    }
  }
}

// ---------------- deconv 2x2 (one (a,b) sub-position per launch) ----------------
// act: f16 NHWC [Nb*Hin*Win][Cin]; wt: f16 [D][C] for this (a,b); acc: f32 [Nb*2Hin*2Win][Cout]
// wave tile: 32 input pixels x 64 out channels (grid.y = out-channel tile of 64)
__global__ void __launch_bounds__(256) k_deconv_wmma(
    const _Float16* __restrict__ act, const _Float16* __restrict__ wt,
    float* __restrict__ acc,
    int Nb, int Hin, int Win, int Cin, int Cout, int a, int b)
{
  const int lane = threadIdx.x & 31;
  const int wave = threadIdx.x >> 5;
  const unsigned numM = ((unsigned)(Nb * Hin * Win)) >> 5;
  const unsigned mt = blockIdx.x * 8u + (unsigned)wave;
  if (mt >= numM) return;
  const int nb = (int)(blockIdx.y << 6);

  const int col  = lane & 15;
  const int khbA = (lane & 16) ? 8 : 0;
  const int kbB  = (lane & 16) ? 8 : 0;

  const unsigned pix0 = mt * 32u + (unsigned)col;
  const _Float16* ap0 = act + (size_t)pix0 * Cin + khbA;
  const _Float16* ap1 = act + (size_t)(pix0 + 16u) * Cin + khbA;
  const _Float16* wp  = wt + ((size_t)(nb + col)) * Cin + kbB;

  const v8f zf = {0.f,0.f,0.f,0.f,0.f,0.f,0.f,0.f};
  v8f cacc[2][4];
#pragma unroll
  for (int g = 0; g < 2; ++g)
#pragma unroll
    for (int j = 0; j < 4; ++j) cacc[g][j] = zf;

  const int ksteps = Cin >> 5;
  for (int kk = 0; kk < ksteps; ++kk) {
    const int k0 = kk << 5;
    AF a0, a1;
    a0.h[0] = *(const v8h*)(ap0 + k0); a0.h[1] = *(const v8h*)(ap0 + k0 + 16);
    a1.h[0] = *(const v8h*)(ap1 + k0); a1.h[1] = *(const v8h*)(ap1 + k0 + 16);
    AF b4[4];
#pragma unroll
    for (int j = 0; j < 4; ++j) {
      const _Float16* w0 = wp + (size_t)(j * 16) * Cin + k0;
      b4[j].h[0] = *(const v8h*)(w0);
      b4[j].h[1] = *(const v8h*)(w0 + 16);
    }
#pragma unroll
    for (int j = 0; j < 4; ++j) {
      cacc[0][j] = __builtin_amdgcn_wmma_f32_16x16x32_f16(false, a0.v, false, b4[j].v, (short)0, cacc[0][j], false, false);
      cacc[1][j] = __builtin_amdgcn_wmma_f32_16x16x32_f16(false, a1.v, false, b4[j].v, (short)0, cacc[1][j], false, false);
    }
  }
  const int chBase = nb + col;
  const int mAdd = (lane & 16) ? 8 : 0;
  const unsigned HW = (unsigned)(Hin * Win);
#pragma unroll
  for (int g = 0; g < 2; ++g) {
#pragma unroll
    for (int vv = 0; vv < 8; ++vv) {
      const unsigned pi = mt * 32u + (unsigned)(g * 16 + vv + mAdd);
      const unsigned n = pi / HW; const unsigned r = pi % HW;
      const unsigned h = r / (unsigned)Win; const unsigned w = r % (unsigned)Win;
      const size_t po = ((size_t)n * 2 * Hin + 2 * h + a) * (size_t)(2 * Win) + 2 * w + b;
      const size_t o = po * Cout + chBase;
      acc[o]      = cacc[g][0][vv];
      acc[o + 16] = cacc[g][1][vv];
      acc[o + 32] = cacc[g][2][vv];
      acc[o + 48] = cacc[g][3][vv];
    }
  }
}

// ---------------- per-pixel LayerNorm epilogue ----------------
// mode 0: out16 = relu(ln(acc+bias)*g+be)
// mode 1: fused32 = (level==0 ? 0 : fused32) + fw[level]*relu(ln(...))
// mode 2: out16 = relu(xres + ln(...))
__global__ void __launch_bounds__(256) k_ln_epilogue(
    const float* __restrict__ acc, const float* __restrict__ bias,
    const float* __restrict__ gamma, const float* __restrict__ beta,
    _Float16* __restrict__ out16, float* __restrict__ fused32,
    const _Float16* __restrict__ xres, const float* __restrict__ fwv,
    int P, int C, int mode, int level)
{
  int w = (int)((blockIdx.x * blockDim.x + threadIdx.x) >> 5);
  int lane = threadIdx.x & 31;
  if (w >= P) return;
  const float* row = acc + (size_t)w * C;
  const int cpl = C >> 5;                 // 8 / 4 / 2
  float v[8];
  float s = 0.f, s2 = 0.f;
#pragma unroll
  for (int i = 0; i < 8; ++i) if (i < cpl) {
    int c = i * 32 + lane;
    float x = row[c] + bias[c];
    v[i] = x; s += x; s2 += x * x;
  }
#pragma unroll
  for (int off = 16; off > 0; off >>= 1) {
    s  += __shfl_xor(s,  off, 32);
    s2 += __shfl_xor(s2, off, 32);
  }
  const float invC = 1.f / (float)C;
  const float m  = s * invC;
  const float var = s2 * invC - m * m;
  const float rs = rsqrtf(var + 1e-6f);
  const float fw = (mode == 1) ? fwv[level] : 0.f;
#pragma unroll
  for (int i = 0; i < 8; ++i) if (i < cpl) {
    int c = i * 32 + lane;
    float y = (v[i] - m) * rs * gamma[c] + beta[c];
    size_t o = (size_t)w * C + c;
    if (mode == 0) {
      out16[o] = (_Float16)fmaxf(y, 0.f);
    } else if (mode == 1) {
      float r = fw * fmaxf(y, 0.f);
      fused32[o] = (level == 0) ? r : (fused32[o] + r);
    } else {
      float r = (float)xres[o] + y;
      out16[o] = (_Float16)fmaxf(r, 0.f);
    }
  }
}

// ---------------- final 1x1 head (64 -> 3), NHWC f16 -> NCHW f32 ----------------
__global__ void k_out_head(const _Float16* __restrict__ act, const float* __restrict__ ow,
                           const float* __restrict__ ob, float* __restrict__ out, int P)
{
  int p = (int)(blockIdx.x * 256 + threadIdx.x);
  if (p >= P) return;
  float a0 = ob[0], a1 = ob[1], a2 = ob[2];
  const _Float16* r = act + (size_t)p * 64;
  for (int c = 0; c < 64; ++c) {
    float a = (float)r[c];
    a0 += a * ow[c]; a1 += a * ow[64 + c]; a2 += a * ow[128 + c];
  }
  unsigned n = (unsigned)p / 3136u; unsigned rr = (unsigned)p % 3136u;   // 56*56
  out[((size_t)n * 3 + 0) * 3136 + rr] = a0;
  out[((size_t)n * 3 + 1) * 3136 + rr] = a1;
  out[((size_t)n * 3 + 2) * 3136 + rr] = a2;
}

// ---------------- host ----------------
static inline dim3 conv_grid(int Pout, int Cout) {
  unsigned numM = (unsigned)(Pout >> 5);        // 32-pixel tiles
  return dim3((numM + 7) / 8, (unsigned)(Cout >> 6), 1);
}

extern "C" void kernel_launch(void* const* d_in, const int* in_sizes, int n_in,
                              void* d_out, int out_size, void* d_ws, size_t ws_size,
                              hipStream_t stream) {
  (void)in_sizes; (void)out_size; (void)ws_size;
  const float* F[64];
  for (int i = 0; i < n_in && i < 64; ++i) F[i] = (const float*)d_in[i];

  char* ws = (char*)d_ws;
  float*    ACC   = (float*)(ws + OFF_ACC);
  _Float16* X0    = (_Float16*)(ws + OFF_X0);
  _Float16* X1    = (_Float16*)(ws + OFF_X1);
  float*    FUSED = (float*)(ws + OFF_FUSED);
  _Float16* W16   = (_Float16*)(ws + OFF_W);
  float*    FW    = (float*)(ws + OFF_FW);

  // ---- weight prep (f32 -> f16, repacked) ----
  for (int l = 0; l < 4; ++l)
    k_f32_to_f16<<<(65536 + 255) / 256, 256, 0, stream>>>(F[51 + 4 * l], W16 + WOFF_RED + (size_t)l * 65536, 65536);
  const int resbase[3] = {19, 27, 35};   // res1/res2/res3 param base indices
  size_t wro = WOFF_RES;
  for (int j = 0; j < 3; ++j) {
    k_oihw_to_toc<<<(589824 + 255) / 256, 256, 0, stream>>>(F[resbase[j] + 6], W16 + wro, 256, 256, 9); wro += 589824;
    k_oihw_to_toc<<<(589824 + 255) / 256, 256, 0, stream>>>(F[resbase[j] + 7], W16 + wro, 256, 256, 9); wro += 589824;
  }
  k_cdab_to_abdc<<<(262144 + 255) / 256, 256, 0, stream>>>(F[44], W16 + WOFF_UP1, 256, 256);
  k_cdab_to_abdc<<<(131072 + 255) / 256, 256, 0, stream>>>(F[46], W16 + WOFF_UP2, 256, 128);
  k_oihw_to_toc<<<(147456 + 255) / 256, 256, 0, stream>>>(F[6], W16 + WOFF_C1, 128, 128, 9);
  k_oihw_to_toc<<<(73728 + 255) / 256, 256, 0, stream>>>(F[8], W16 + WOFF_C2, 64, 128, 9);
  k_softmax4<<<1, 32, 0, stream>>>(F[47], FW);

  // ---- per-level: ROI align -> 1x1 conv -> LN+ReLU -> weighted fuse ----
  const int   Hs[4] = {256, 128, 64, 32};
  const float sc[4] = {0.25f, 0.125f, 0.0625f, 0.03125f};
  for (int l = 0; l < 4; ++l) {
    k_roi_align<<<P28, 256, 0, stream>>>(F[l], F[4], X0, 256, Hs[l], Hs[l], sc[l]);
    k_conv_wmma<<<conv_grid(P28, 256), 256, 0, stream>>>(X0, W16 + WOFF_RED + (size_t)l * 65536, ACC,
                                                         48, 28, 28, 256, 28, 28, 256, 1, 1, 0);
    k_ln_epilogue<<<P28 / 8, 256, 0, stream>>>(ACC, F[48 + 4 * l], F[50 + 4 * l], F[49 + 4 * l],
                                               nullptr, FUSED, nullptr, FW, P28, 256, 1, l);
  }
  k_f32_to_f16<<<(P28 * 256 + 255) / 256, 256, 0, stream>>>(FUSED, X0, P28 * 256);

  // ---- res1 (28x28, 256ch), X = X0 ----
  k_conv_wmma<<<conv_grid(P28, 256), 256, 0, stream>>>(X0, W16 + WOFF_RES + (size_t)0 * 589824, ACC,
                                                       48, 28, 28, 256, 28, 28, 256, 3, 1, 1);
  k_ln_epilogue<<<P28 / 8, 256, 0, stream>>>(ACC, F[19], F[23], F[21], X1, nullptr, nullptr, nullptr, P28, 256, 0, 0);
  k_conv_wmma<<<conv_grid(P28, 256), 256, 0, stream>>>(X1, W16 + WOFF_RES + (size_t)1 * 589824, ACC,
                                                       48, 28, 28, 256, 28, 28, 256, 3, 1, 1);
  k_ln_epilogue<<<P28 / 8, 256, 0, stream>>>(ACC, F[20], F[24], F[22], X1, nullptr, X0, nullptr, P28, 256, 2, 0);

  // ---- res2, X = X1 ----
  k_conv_wmma<<<conv_grid(P28, 256), 256, 0, stream>>>(X1, W16 + WOFF_RES + (size_t)2 * 589824, ACC,
                                                       48, 28, 28, 256, 28, 28, 256, 3, 1, 1);
  k_ln_epilogue<<<P28 / 8, 256, 0, stream>>>(ACC, F[27], F[31], F[29], X0, nullptr, nullptr, nullptr, P28, 256, 0, 0);
  k_conv_wmma<<<conv_grid(P28, 256), 256, 0, stream>>>(X0, W16 + WOFF_RES + (size_t)3 * 589824, ACC,
                                                       48, 28, 28, 256, 28, 28, 256, 3, 1, 1);
  k_ln_epilogue<<<P28 / 8, 256, 0, stream>>>(ACC, F[28], F[32], F[30], X0, nullptr, X1, nullptr, P28, 256, 2, 0);

  // ---- up1: deconv 2x2 (256->256, 28->56) + LN+ReLU ----
  for (int ab = 0; ab < 4; ++ab)
    k_deconv_wmma<<<conv_grid(P28, 256), 256, 0, stream>>>(X0, W16 + WOFF_UP1 + (size_t)ab * 256 * 256, ACC,
                                                           48, 28, 28, 256, 256, ab >> 1, ab & 1);
  k_ln_epilogue<<<P56 / 8, 256, 0, stream>>>(ACC, F[43], F[10], F[9], X1, nullptr, nullptr, nullptr, P56, 256, 0, 0);

  // ---- res3 (56x56, 256ch), X = X1 ----
  k_conv_wmma<<<conv_grid(P56, 256), 256, 0, stream>>>(X1, W16 + WOFF_RES + (size_t)4 * 589824, ACC,
                                                       48, 56, 56, 256, 56, 56, 256, 3, 1, 1);
  k_ln_epilogue<<<P56 / 8, 256, 0, stream>>>(ACC, F[35], F[39], F[37], X0, nullptr, nullptr, nullptr, P56, 256, 0, 0);
  k_conv_wmma<<<conv_grid(P56, 256), 256, 0, stream>>>(X0, W16 + WOFF_RES + (size_t)5 * 589824, ACC,
                                                       48, 56, 56, 256, 56, 56, 256, 3, 1, 1);
  k_ln_epilogue<<<P56 / 8, 256, 0, stream>>>(ACC, F[36], F[40], F[38], X0, nullptr, X1, nullptr, P56, 256, 2, 0);

  // ---- up2: deconv 2x2 (256->128, 56->112) + LN+ReLU ----
  for (int ab = 0; ab < 4; ++ab)
    k_deconv_wmma<<<conv_grid(P56, 128), 256, 0, stream>>>(X0, W16 + WOFF_UP2 + (size_t)ab * 128 * 256, ACC,
                                                           48, 56, 56, 256, 128, ab >> 1, ab & 1);
  k_ln_epilogue<<<P112 / 8, 256, 0, stream>>>(ACC, F[45], F[12], F[11], X1, nullptr, nullptr, nullptr, P112, 128, 0, 0);

  // ---- c1: 3x3 (128->128, 112x112) + LN+ReLU ----
  k_conv_wmma<<<conv_grid(P112, 128), 256, 0, stream>>>(X1, W16 + WOFF_C1, ACC,
                                                        48, 112, 112, 128, 112, 112, 128, 3, 1, 1);
  k_ln_epilogue<<<P112 / 8, 256, 0, stream>>>(ACC, F[5], F[14], F[13], X0, nullptr, nullptr, nullptr, P112, 128, 0, 0);

  // ---- c2: 3x3 stride-2 (128->64, 112->56) + LN+ReLU ----
  k_conv_wmma<<<conv_grid(P56, 64), 256, 0, stream>>>(X0, W16 + WOFF_C2, ACC,
                                                      48, 112, 112, 128, 56, 56, 64, 3, 2, 1);
  k_ln_epilogue<<<P56 / 8, 256, 0, stream>>>(ACC, F[7], F[16], F[15], X1, nullptr, nullptr, nullptr, P56, 64, 0, 0);

  // ---- output head: 1x1 (64 -> 3), NCHW f32 ----
  k_out_head<<<(P56 + 255) / 256, 256, 0, stream>>>(X1, F[18], F[17], (float*)d_out, P56);
}
